// GCN_AUG_52527450030632
// MI455X (gfx1250) — compile-verified, matrix operand order checked by reading, not run on hardware
//
#include <hip/hip_runtime.h>
#include <cstdint>

// ---------------- problem constants ----------------
#define NNODE 6144
#define FIN   256
#define HID   64
#define NEDGE 196608
#define ALPHA 0.8f
#define EPSP  1e-6f
#define BNEPS 1e-5f

typedef __attribute__((ext_vector_type(16))) __bf16 v16bf;
typedef __attribute__((ext_vector_type(8)))  __bf16 v8bf;
typedef __attribute__((ext_vector_type(8)))  float  v8f;
typedef __attribute__((ext_vector_type(4)))  float  v4f;
typedef __attribute__((ext_vector_type(4)))  uint32_t u32x4;
typedef __attribute__((ext_vector_type(8)))  uint32_t u32x8;

// -------- CDNA5 async copy: global -> LDS, 16 bytes per lane --------
__device__ __forceinline__ void async_copy_b128(void* lds, const void* gptr) {
  unsigned l = (unsigned)(uintptr_t)lds;
  asm volatile("global_load_async_to_lds_b128 %0, %1, off"
               :: "v"(l), "v"(gptr)
               : "memory");
}
__device__ __forceinline__ void async_wait0() {
  asm volatile("s_wait_asynccnt 0x0" ::: "memory");
}

// -------- CDNA5 Tensor Data Mover: 2D tile load, D# groups 0/1 in SGPRs ----
__device__ __forceinline__ void tdm_load_2d(u32x4 g0, u32x8 g1) {
  asm volatile("tensor_load_to_lds %0, %1" :: "s"(g0), "s"(g1) : "memory");
}

// Build D# group0: count=1, lds byte addr, 57-bit global addr, type=2
__device__ __forceinline__ u32x4 tdm_group0(void* lds, const void* g) {
  uint64_t ga = (uint64_t)(uintptr_t)g;
  u32x4 r;
  r[0] = 1u;                                  // count=1, user descriptor
  r[1] = (uint32_t)(uintptr_t)lds;            // lds_addr
  r[2] = (uint32_t)ga;                        // global_addr[31:0]
  r[3] = (uint32_t)((ga >> 32) & 0x01FFFFFFu) // global_addr[56:32]
         | (2u << 30);                        // type=2 ("image")
  return r;
}

// ---- D# group1 for a 128-row x 128-byte tile out of a row-major tensor ----
// data_size=4B (32 dwords/row), LDS pad 16B per 32 dwords -> 144B LDS rows.
// tensor_dim0=32 dw, tensor_dim1=6144 rows, tile 32 dw x 128 rows, stride 32 dw.
__device__ __forceinline__ u32x8 tdm_group1_tile128() {
  u32x8 r;
  r[0] = (2u << 16)        // data_size = 4 bytes
       | (1u << 20)        // pad_enable
       | (4u << 22)        // pad_interval: 32 dwords
       | (3u << 25);       // pad_amount: 4 dwords (16B)
  r[1] = 32u << 16;        // tensor_dim0[15:0] = 32
  r[2] = 6144u << 16;      // tensor_dim1[15:0] = 6144
  r[3] = 32u << 16;        // tile_dim0 = 32
  r[4] = 128u;             // tile_dim1 = 128
  r[5] = 32u;              // tensor_dim0_stride = 32 dwords
  r[6] = 0u;
  r[7] = 0u;
  return r;
}

// -------- WMMA fragment helpers (CDNA5 16x16x32 bf16, wave32) --------
__device__ __forceinline__ v16bf load_frag(const __bf16* __restrict__ base, int ld) {
  const int lane = threadIdx.x & 31;
  const int r    = lane & 15;
  const int hi   = lane >> 4;
  const __bf16* p = base + r * ld + hi * 8;
  union { v16bf v; v8bf h[2]; } u;
  u.h[0] = *(const v8bf*)(p);
  u.h[1] = *(const v8bf*)(p + 16);
  return u.v;
}

__device__ __forceinline__ void store_acc(float* __restrict__ out, int ld, v8f a) {
  const int lane = threadIdx.x & 31;
  const int c    = lane & 15;
  const int hi   = lane >> 4;
#pragma unroll
  for (int i = 0; i < 8; ++i) out[(i + 8 * hi) * ld + c] = a[i];
}

// ---------------- elementwise / setup kernels ----------------
__global__ void k_f2bf(const float* __restrict__ s, __bf16* __restrict__ d, int n) {
  for (int i = blockIdx.x * 256 + threadIdx.x; i < n; i += gridDim.x * 256)
    d[i] = (__bf16)s[i];
}

// transpose-convert: src[K x 64] f32 -> dst[64 x K] bf16
__global__ void k_f2bf_T(const float* __restrict__ s, __bf16* __restrict__ d, int K) {
  int idx = blockIdx.x * 256 + threadIdx.x;  // 64*K threads
  int n = idx / K, k = idx - n * K;
  d[idx] = (__bf16)s[k * 64 + n];
}

__global__ void k_zero4(float* __restrict__ p, int n4) {
  v4f z = {0.f, 0.f, 0.f, 0.f};
  for (int i = blockIdx.x * 256 + threadIdx.x; i < n4; i += gridDim.x * 256)
    ((v4f*)p)[i] = z;
}

__global__ void k_deg(const long long* __restrict__ ei, float* __restrict__ deg) {
  int e = blockIdx.x * 256 + threadIdx.x;
  if (e < NEDGE) atomicAdd(&deg[(int)ei[NEDGE + e]], 1.0f);
}

__global__ void k_dinv0(const float* __restrict__ deg, float* __restrict__ dinv0) {
  int i = blockIdx.x * 256 + threadIdx.x;
  if (i < NNODE) dinv0[i] = rsqrtf(deg[i] + 1.0f);  // +1 self loop
}

// ---------------- skinny GEMM: C[M x 64] = A[M x K] * Bt^T, Bt[64 x K] ----------------
__global__ void __launch_bounds__(256) k_gemm_skinny(
    const __bf16* __restrict__ A, const __bf16* __restrict__ Bt,
    float* __restrict__ C, int K) {
  __shared__ __bf16 As[128 * 72];
  __shared__ __bf16 Bs[64 * 72];
  const int tid  = threadIdx.x;
  const int wave = tid >> 5;
  const int row0 = blockIdx.x * 128;
  v8f acc[4] = {};

  for (int kc = 0; kc < K; kc += 64) {
    for (int idx = tid; idx < 128 * 8; idx += 256) {          // A tile 128x64
      int r = idx >> 3, c8 = (idx & 7) * 8;
      async_copy_b128(&As[r * 72 + c8], &A[(row0 + r) * K + kc + c8]);
    }
    for (int idx = tid; idx < 64 * 8; idx += 256) {           // Bt tile 64x64
      int n = idx >> 3, k8 = (idx & 7) * 8;
      async_copy_b128(&Bs[n * 72 + k8], &Bt[n * K + kc + k8]);
    }
    async_wait0();
    __syncthreads();
#pragma unroll
    for (int ks = 0; ks < 64; ks += 32) {
      v16bf a = load_frag(&As[wave * 16 * 72 + ks], 72);
#pragma unroll
      for (int ct = 0; ct < 4; ++ct) {
        v16bf b = load_frag(&Bs[ct * 16 * 72 + ks], 72);
        acc[ct] = __builtin_amdgcn_wmma_f32_16x16x32_bf16(
            false, a, false, b, (short)0, acc[ct], false, false);
      }
    }
    __syncthreads();
  }
#pragma unroll
  for (int ct = 0; ct < 4; ++ct)
    store_acc(&C[(row0 + wave * 16) * 64 + ct * 16], 64, acc[ct]);
}

// ---------------- edge-space GCN conv ----------------
__global__ void k_edge_msg(const long long* __restrict__ ei, const float* __restrict__ h0,
                           const float* __restrict__ dinv0, float* __restrict__ h) {
  int e = blockIdx.x * 4 + (threadIdx.x >> 6);
  int f = threadIdx.x & 63;
  if (e >= NEDGE) return;
  int s = (int)ei[e], d = (int)ei[NEDGE + e];
  float nrm = dinv0[s] * dinv0[d];
  atomicAdd(&h[d * 64 + f], h0[s * 64 + f] * nrm);
}

__global__ void k_self_bias(const float* __restrict__ h0, const float* __restrict__ dinv0,
                            const float* __restrict__ b1, float* __restrict__ h,
                            __bf16* __restrict__ hbf) {
  int idx = blockIdx.x * 256 + threadIdx.x;    // N*64 threads
  int i = idx >> 6, f = idx & 63;
  float di = dinv0[i];
  float v = h[idx] + h0[idx] * di * di + b1[f];
  h[idx] = v;
  hbf[idx] = (__bf16)v;
}

// ---------------- adj_logits = h @ h^T  (128x128 tile / block, TDM-fed) ------
__global__ void __launch_bounds__(256) k_gemm_logits(const __bf16* __restrict__ H,
                                                     float* __restrict__ out) {
  __shared__ __bf16 As[128 * 72];
  __shared__ __bf16 Bs[128 * 72];
  const int tid  = threadIdx.x;
  const int wave = tid >> 5;
  const int row0 = blockIdx.y * 128, col0 = blockIdx.x * 128;

  if (wave == 0) {  // one wave issues both tile DMAs (TDM, TENSORcnt-tracked)
    u32x8 g1 = tdm_group1_tile128();
    tdm_load_2d(tdm_group0(&As[0], H + (size_t)row0 * 64), g1);
    tdm_load_2d(tdm_group0(&Bs[0], H + (size_t)col0 * 64), g1);
    __builtin_amdgcn_s_wait_tensorcnt(0);
  }
  __syncthreads();

  v8f acc[8] = {};
#pragma unroll
  for (int ks = 0; ks < 64; ks += 32) {
    v16bf a = load_frag(&As[wave * 16 * 72 + ks], 72);
#pragma unroll
    for (int ct = 0; ct < 8; ++ct) {
      v16bf b = load_frag(&Bs[ct * 16 * 72 + ks], 72);
      acc[ct] = __builtin_amdgcn_wmma_f32_16x16x32_bf16(
          false, a, false, b, (short)0, acc[ct], false, false);
    }
  }
#pragma unroll
  for (int ct = 0; ct < 8; ++ct)
    store_acc(&out[(size_t)(row0 + wave * 16) * NNODE + col0 + ct * 16], NNODE, acc[ct]);
}

// ---------------- max reduction ----------------
__global__ void k_max_red(const float* __restrict__ x, int n, float* __restrict__ out) {
  __shared__ float sm[256];
  float m = -3.4e38f;
  for (int i = blockIdx.x * 256 + threadIdx.x; i < n; i += gridDim.x * 256)
    m = fmaxf(m, x[i]);
  sm[threadIdx.x] = m;
  __syncthreads();
  for (int s = 128; s > 0; s >>= 1) {
    if (threadIdx.x < s) sm[threadIdx.x] = fmaxf(sm[threadIdx.x], sm[threadIdx.x + s]);
    __syncthreads();
  }
  if (!threadIdx.x) out[blockIdx.x] = sm[0];
}

__global__ void k_orig(const long long* __restrict__ ei, float* __restrict__ orig) {
  int e = blockIdx.x * 256 + threadIdx.x;
  if (e < NEDGE) orig[(size_t)(int)ei[e] * NNODE + (int)ei[NEDGE + e]] = 1.0f;
}

// ---------------- relaxed-Bernoulli sample + symmetrize ----------------
__global__ void k_sample(const float* __restrict__ lg, const float* __restrict__ og,
                         const float* __restrict__ u, const float* __restrict__ maxv,
                         float* __restrict__ adjs) {
  int idx = blockIdx.x * 256 + threadIdx.x;  // N*N threads
  int i = idx / NNODE, j = idx - i * NNODE;
  if (i == j) { adjs[idx] = 1.0f; return; }
  int r = i < j ? i : j, c = i < j ? j : i;  // upper-triangle source
  int o = r * NNODE + c;
  float p = ALPHA * (lg[o] / maxv[0]) + (1.0f - ALPHA) * og[o];
  p = fmaxf(p, 0.0f);
  p = fminf(fmaxf(p, EPSP), 1.0f - EPSP);
  float lt = logf(p) - log1pf(-p);
  float uu = u[o];
  float nz = logf(uu) - log1pf(-uu);
  float soft = 1.0f / (1.0f + expf(-(lt + nz)));
  adjs[idx] = rintf(soft);  // straight-through hard sample (forward value)
}

__global__ void k_rowsum(const float* __restrict__ adjs, float* __restrict__ dinvS) {
  __shared__ float sm[256];
  const float* row = adjs + (size_t)blockIdx.x * NNODE;
  float s = 0.f;
  for (int j = threadIdx.x; j < NNODE; j += 256) s += row[j];
  sm[threadIdx.x] = s;
  __syncthreads();
  for (int t = 128; t > 0; t >>= 1) {
    if (threadIdx.x < t) sm[threadIdx.x] += sm[threadIdx.x + t];
    __syncthreads();
  }
  if (!threadIdx.x) dinvS[blockIdx.x] = rsqrtf(sm[0] + 1.0f);  // A = adjS + I
}

// vT[f][i] = dinvS[i] * src[i][f]   (global transpose so GEMM B-fills are copies)
__global__ void k_prep_vT(const float* __restrict__ src, const float* __restrict__ dinvS,
                          __bf16* __restrict__ vt) {
  int idx = blockIdx.x * 256 + threadIdx.x;  // 64*N threads
  int f = idx / NNODE, i = idx - f * NNODE;
  vt[idx] = (__bf16)(dinvS[i] * src[i * 64 + f]);
}

// ---------------- S = adjS(f32, N x N) @ V, with Vt[64 x N] bf16 ----------------
__global__ void __launch_bounds__(256) k_gemm_av(const float* __restrict__ A,
                                                 const __bf16* __restrict__ Vtg,
                                                 float* __restrict__ S) {
  __shared__ __bf16 As[128 * 72];
  __shared__ __bf16 Vt[64 * 72];
  const int tid  = threadIdx.x;
  const int wave = tid >> 5;
  const int row0 = blockIdx.x * 128;
  v8f acc[4] = {};
  for (int kc = 0; kc < NNODE; kc += 64) {
    for (int idx = tid; idx < 128 * 16; idx += 256) {  // A tile 128x64, f32 -> bf16
      int r = idx >> 4, c4 = (idx & 15) * 4;
      const float* gp = &A[(size_t)(row0 + r) * NNODE + kc + c4];
      v4f f = *(const v4f*)gp;
      if (kc + 64 < NNODE) __builtin_prefetch(gp + 64, 0, 1);  // next K chunk
      As[r * 72 + c4 + 0] = (__bf16)f[0];
      As[r * 72 + c4 + 1] = (__bf16)f[1];
      As[r * 72 + c4 + 2] = (__bf16)f[2];
      As[r * 72 + c4 + 3] = (__bf16)f[3];
    }
    for (int idx = tid; idx < 64 * 8; idx += 256) {  // Vt tile: straight async copy
      int n = idx >> 3, k8 = (idx & 7) * 8;
      async_copy_b128(&Vt[n * 72 + k8], &Vtg[n * NNODE + kc + k8]);
    }
    async_wait0();
    __syncthreads();
#pragma unroll
    for (int ks = 0; ks < 64; ks += 32) {
      v16bf a = load_frag(&As[wave * 16 * 72 + ks], 72);
#pragma unroll
      for (int ct = 0; ct < 4; ++ct) {
        v16bf b = load_frag(&Vt[ct * 16 * 72 + ks], 72);
        acc[ct] = __builtin_amdgcn_wmma_f32_16x16x32_bf16(
            false, a, false, b, (short)0, acc[ct], false, false);
      }
    }
    __syncthreads();
  }
#pragma unroll
  for (int ct = 0; ct < 4; ++ct)
    store_acc(&S[(row0 + wave * 16) * 64 + ct * 16], 64, acc[ct]);
}

// ---------------- tails ----------------
__global__ void k_finish_x1(const float* __restrict__ S, const float* __restrict__ h0,
                            const float* __restrict__ dinvS, const float* __restrict__ b1,
                            const float* __restrict__ prelu_a, float* __restrict__ x1,
                            float* __restrict__ z) {
  int idx = blockIdx.x * 256 + threadIdx.x;
  int i = idx >> 6, f = idx & 63;
  float di = dinvS[i];
  float v = di * S[idx] + di * di * h0[idx] + b1[f];
  x1[idx] = v;
  float a = prelu_a[0];
  z[idx] = v >= 0.f ? v : a * v;
}

__global__ void k_bn_stats(const float* __restrict__ x1, float* __restrict__ bn) {
  __shared__ float s1[256], s2[256];
  int f = blockIdx.x;
  float a = 0.f, b = 0.f;
  for (int i = threadIdx.x; i < NNODE; i += 256) {
    float v = x1[i * 64 + f];
    a += v; b += v * v;
  }
  s1[threadIdx.x] = a; s2[threadIdx.x] = b;
  __syncthreads();
  for (int t = 128; t > 0; t >>= 1) {
    if (threadIdx.x < t) { s1[threadIdx.x] += s1[threadIdx.x + t]; s2[threadIdx.x] += s2[threadIdx.x + t]; }
    __syncthreads();
  }
  if (!threadIdx.x) {
    float mean = s1[0] / (float)NNODE;
    float var  = s2[0] / (float)NNODE - mean * mean;  // biased
    bn[f] = mean;
    bn[64 + f] = rsqrtf(var + BNEPS);
  }
}

__global__ void k_bn_apply(const float* __restrict__ x1, const float* __restrict__ bn,
                           const float* __restrict__ gamma, const float* __restrict__ beta,
                           __bf16* __restrict__ xrbf) {
  int idx = blockIdx.x * 256 + threadIdx.x;
  int f = idx & 63;
  float v = gamma[f] * (x1[idx] - bn[f]) * bn[64 + f] + beta[f];
  xrbf[idx] = (__bf16)fmaxf(v, 0.f);
}

__global__ void k_finish_x2(const float* __restrict__ S, const float* __restrict__ g,
                            const float* __restrict__ dinvS, const float* __restrict__ b2,
                            float* __restrict__ x2) {
  int idx = blockIdx.x * 256 + threadIdx.x;
  int i = idx >> 6, f = idx & 63;
  float di = dinvS[i];
  x2[idx] = di * S[idx] + di * di * g[idx] + b2[f];
}

// ---------------- host launcher ----------------
static inline size_t alup(size_t x) { return (x + 255) & ~(size_t)255; }

extern "C" void kernel_launch(void* const* d_in, const int* in_sizes, int n_in,
                              void* d_out, int out_size, void* d_ws, size_t ws_size,
                              hipStream_t stream) {
  (void)in_sizes; (void)n_in; (void)out_size; (void)ws_size;
  const float*     x   = (const float*)d_in[0];
  const long long* ei  = (const long long*)d_in[1];   // int64 edge_index [2,E]
  const float*     W1  = (const float*)d_in[2];
  const float*     b1  = (const float*)d_in[3];
  const float*     W2  = (const float*)d_in[4];
  const float*     b2  = (const float*)d_in[5];
  const float*     gm  = (const float*)d_in[6];
  const float*     bt  = (const float*)d_in[7];
  const float*     pa  = (const float*)d_in[8];
  const float*     u   = (const float*)d_in[9];

  // outputs: (x2, z, adj_sampled, adj_logits, adj_orig) flat-concatenated
  float* out    = (float*)d_out;
  float* o_x2   = out;
  float* o_z    = out + (size_t)NNODE * HID;
  float* o_adjs = o_z + (size_t)NNODE * HID;
  float* o_lg   = o_adjs + (size_t)NNODE * NNODE;
  float* o_og   = o_lg + (size_t)NNODE * NNODE;

  // workspace bump allocation
  char* w = (char*)d_ws;
  size_t off = 0;
  auto alloc = [&](size_t bytes) { void* p = w + off; off += alup(bytes); return p; };
  __bf16* x_bf   = (__bf16*)alloc((size_t)NNODE * FIN * 2);
  __bf16* W1t_bf = (__bf16*)alloc((size_t)HID * FIN * 2);   // [64][256]
  __bf16* W2t_bf = (__bf16*)alloc((size_t)HID * HID * 2);   // [64][64]
  float*  h0     = (float*)alloc((size_t)NNODE * HID * 4);
  float*  h      = (float*)alloc((size_t)NNODE * HID * 4);
  __bf16* h_bf   = (__bf16*)alloc((size_t)NNODE * HID * 2);
  float*  deg    = (float*)alloc((size_t)NNODE * 4);
  float*  dinv0  = (float*)alloc((size_t)NNODE * 4);
  float*  dinvS  = (float*)alloc((size_t)NNODE * 4);
  float*  part   = (float*)alloc(1024 * 4);
  float*  maxv   = (float*)alloc(256);
  __bf16* vt_bf  = (__bf16*)alloc((size_t)HID * NNODE * 2); // [64][N]
  float*  Sbuf   = (float*)alloc((size_t)NNODE * HID * 4);
  float*  x1     = (float*)alloc((size_t)NNODE * HID * 4);
  float*  g      = (float*)alloc((size_t)NNODE * HID * 4);
  __bf16* xr_bf  = (__bf16*)alloc((size_t)NNODE * HID * 2);
  float*  bn     = (float*)alloc(128 * 4);

  const int NF = NNODE * HID;  // 393216

  // 1) input conversions f32 -> bf16 (weights transposed for async B-fills)
  k_f2bf<<<1024, 256, 0, stream>>>(x, x_bf, NNODE * FIN);
  k_f2bf_T<<<(HID * FIN) / 256, 256, 0, stream>>>(W1, W1t_bf, FIN);
  k_f2bf_T<<<(HID * HID) / 256, 256, 0, stream>>>(W2, W2t_bf, HID);

  // 2) zero: deg, h accumulator, adj_orig output
  k_zero4<<<6, 256, 0, stream>>>(deg, NNODE / 4);
  k_zero4<<<384, 256, 0, stream>>>(h, NF / 4);
  k_zero4<<<36864, 256, 0, stream>>>(o_og, (int)(((size_t)NNODE * NNODE) / 4));

  // 3) original-graph degrees + D^-1/2
  k_deg<<<NEDGE / 256, 256, 0, stream>>>(ei, deg);
  k_dinv0<<<NNODE / 256, 256, 0, stream>>>(deg, dinv0);

  // 4) h0 = x @ W1  (WMMA, K=256)
  k_gemm_skinny<<<NNODE / 128, 256, 0, stream>>>(x_bf, W1t_bf, h0, FIN);

  // 5) GCN edge conv on original graph -> h, h_bf
  k_edge_msg<<<NEDGE / 4, 256, 0, stream>>>(ei, h0, dinv0, h);
  k_self_bias<<<NF / 256, 256, 0, stream>>>(h0, dinv0, b1, h, h_bf);

  // 6) adj_logits = h @ h^T  (WMMA, N x N, TDM tile feed)
  dim3 glg(NNODE / 128, NNODE / 128);
  k_gemm_logits<<<glg, 256, 0, stream>>>(h_bf, o_lg);

  // 7) global max(adj_logits)
  k_max_red<<<1024, 256, 0, stream>>>(o_lg, (int)((size_t)NNODE * NNODE), part);
  k_max_red<<<1, 256, 0, stream>>>(part, 1024, maxv);

  // 8) adj_orig scatter
  k_orig<<<NEDGE / 256, 256, 0, stream>>>(ei, o_og);

  // 9) sample + symmetrize -> adj_sampled
  k_sample<<<(int)(((size_t)NNODE * NNODE) / 256), 256, 0, stream>>>(o_lg, o_og, u, maxv, o_adjs);

  // 10) sampled-graph D^-1/2
  k_rowsum<<<NNODE, 256, 0, stream>>>(o_adjs, dinvS);

  // 11) x1 = An @ h0 + b1 via S = adjS @ (dinv ⊙ h0)
  k_prep_vT<<<NF / 256, 256, 0, stream>>>(h0, dinvS, vt_bf);
  k_gemm_av<<<NNODE / 128, 256, 0, stream>>>(o_adjs, vt_bf, Sbuf);
  k_finish_x1<<<NF / 256, 256, 0, stream>>>(Sbuf, h0, dinvS, b1, pa, x1, o_z);

  // 12) BN(train) + ReLU -> xr (bf16)
  k_bn_stats<<<HID, 256, 0, stream>>>(x1, bn);
  k_bn_apply<<<NF / 256, 256, 0, stream>>>(x1, bn, gm, bt, xr_bf);

  // 13) g = xr @ W2 (WMMA, K=64)
  k_gemm_skinny<<<NNODE / 128, 256, 0, stream>>>(xr_bf, W2t_bf, g, HID);

  // 14) x2 = An @ g + b2
  k_prep_vT<<<NF / 256, 256, 0, stream>>>(g, dinvS, vt_bf);
  k_gemm_av<<<NNODE / 128, 256, 0, stream>>>(o_adjs, vt_bf, Sbuf);
  k_finish_x2<<<NF / 256, 256, 0, stream>>>(Sbuf, g, dinvS, b2, o_x2);
}
